// myself_multihead_self_attention_4741643895103
// MI455X (gfx1250) — matrix operand order repaired and checked by reading, not compiled, not run on hardware
//
#include <hip/hip_runtime.h>

// ---------------------------------------------------------------------------
// Multi-head self-attention with RoPE for MI455X (gfx1250, wave32, WMMA).
// Pipeline: 3x projection GEMM (f32->f16, RoPE fused for Q/K, V stored
// transposed) -> flash-attention (v_wmma_f32_16x16x32_f16 for QK^T and PV,
// online softmax, async global->LDS staging) -> output projection (f32 out).
// Workspace: 4 x 16MB f16 buffers (Qh, Kh, V^T, ctx) = 64 MB.
// ---------------------------------------------------------------------------

typedef __attribute__((ext_vector_type(16))) _Float16 v16h;
typedef __attribute__((ext_vector_type(8)))  _Float16 v8h;
typedef __attribute__((ext_vector_type(8)))  float    v8f;
typedef __attribute__((ext_vector_type(4)))  float    v4f;
// Pointee type of the async-LDS builtins (from the round-2 diagnostic):
typedef int v4i __attribute__((vector_size(4 * sizeof(int))));

#define D_MODEL 1024
#define NHEAD   16
#define DK      64
#define SEQ     2048
#define BATCH   4
#define MTOT    (BATCH * SEQ)

#if __has_builtin(__builtin_amdgcn_global_load_async_to_lds_b128)
#define HAS_ASYNC_LDS 1
#else
#define HAS_ASYNC_LDS 0
#endif

// Global pointer -> AS(1) v4i*, LDS generic pointer -> AS(3) v4i*
// (flat LDS addresses map to the LDS offset by 32-bit truncation, ISA 10.2).
#define GP1(p) ((__attribute__((address_space(1))) v4i*)(uintptr_t)(p))
#define LP3(p) ((__attribute__((address_space(3))) v4i*)(uint32_t)(uintptr_t)(p))

__device__ __forceinline__ void async_wait0() {
#if __has_builtin(__builtin_amdgcn_s_wait_asynccnt)
  __builtin_amdgcn_s_wait_asynccnt(0);
#else
  asm volatile("s_wait_asynccnt 0" ::: "memory");
#endif
}

// Copy 64 contiguous bytes global -> LDS per lane (ASYNCcnt path if present).
__device__ __forceinline__ void copy64_g2l(const _Float16* g, _Float16* l) {
#if HAS_ASYNC_LDS
  __builtin_amdgcn_global_load_async_to_lds_b128(GP1(g), LP3(l), 0, 0);
  __builtin_amdgcn_global_load_async_to_lds_b128(GP1(g), LP3(l), 16, 0);
  __builtin_amdgcn_global_load_async_to_lds_b128(GP1(g), LP3(l), 32, 0);
  __builtin_amdgcn_global_load_async_to_lds_b128(GP1(g), LP3(l), 48, 0);
#else
  *(v8h*)(l)      = *(const v8h*)(g);
  *(v8h*)(l + 8)  = *(const v8h*)(g + 8);
  *(v8h*)(l + 16) = *(const v8h*)(g + 16);
  *(v8h*)(l + 24) = *(const v8h*)(g + 24);
#endif
}

// Copy 32 contiguous bytes global -> LDS per lane.
__device__ __forceinline__ void copy32_g2l(const _Float16* g, _Float16* l) {
#if HAS_ASYNC_LDS
  __builtin_amdgcn_global_load_async_to_lds_b128(GP1(g), LP3(l), 0, 0);
  __builtin_amdgcn_global_load_async_to_lds_b128(GP1(g), LP3(l), 16, 0);
#else
  *(v8h*)(l)     = *(const v8h*)(g);
  *(v8h*)(l + 8) = *(const v8h*)(g + 8);
#endif
}

__device__ __forceinline__ v8f wmma_f16(v16h a, v16h b, v8f c) {
  // D(16x16,f32) = A(16x32,f16) * B(32x16,f16) + C
  return __builtin_amdgcn_wmma_f32_16x16x32_f16(false, a, false, b,
                                                (short)0, c, false, false);
}

// A-matrix 16x32 f16 fragment gather (ISA 7.12.2): lane m = l&15,
// half j<8 -> K = kb+j, j>=8 -> K = kb+16+(j-8), kb = 8*(l>=16).
__device__ __forceinline__ v16h frag_a16(const _Float16* base, int stride) {
  const int lane = threadIdx.x & 31;
  const _Float16* p = base + (lane & 15) * stride + ((lane >> 4) << 3);
  v8h lo = *(const v8h*)p;
  v8h hi = *(const v8h*)(p + 16);
  v16h r;
#pragma unroll
  for (int i = 0; i < 8; ++i) { r[i] = lo[i]; r[i + 8] = hi[i]; }
  return r;
}

// B-matrix 32x16 f16 fragment gather: lane n = l&15,
// half j<8 -> K = kb+j, j>=8 -> K = kb+16+(j-8), kb = 8*(l>=16).
__device__ __forceinline__ v16h frag_b16(const _Float16* base, int stride) {
  const int lane = threadIdx.x & 31;
  const _Float16* p = base + (lane & 15) * stride + ((lane >> 4) << 3);
  v8h lo = *(const v8h*)p;
  v8h hi = *(const v8h*)(p + 16);
  v16h r;
#pragma unroll
  for (int i = 0; i < 8; ++i) { r[i] = lo[i]; r[i + 8] = hi[i]; }
  return r;
}

// Stage 16 contiguous f32 elements to LDS with f32 -> f16 conversion.
__device__ __forceinline__ void stage16(const float* g, _Float16* l) {
  v4f a0 = *(const v4f*)(g + 0);
  v4f a1 = *(const v4f*)(g + 4);
  v4f a2 = *(const v4f*)(g + 8);
  v4f a3 = *(const v4f*)(g + 12);
  v8h h0, h1;
#pragma unroll
  for (int i = 0; i < 4; ++i) {
    h0[i]     = (_Float16)a0[i];
    h0[i + 4] = (_Float16)a1[i];
    h1[i]     = (_Float16)a2[i];
    h1[i + 4] = (_Float16)a3[i];
  }
  *(v8h*)l = h0;
  *(v8h*)(l + 8) = h1;
}

// ---------------------------------------------------------------------------
// GEMM: out = A[M x 1024] * W[1024 x 1024]^T  (Y[m,n] = sum_k A[m,k]*W[n,k]).
// Block tile 128x128, K-step 32, 8 waves in a 2x4 grid; each wave: 4x2 tiles.
// MODE 0: f16 out, RoPE, head-major [b,h,s,d]      (Q / K projections)
// MODE 1: f16 out, transposed head-major [b,h,d,s] (V projection)
// MODE 2: f32 out, row-major [m,n]                 (output projection)
// ---------------------------------------------------------------------------
template <int MODE, typename TA>
__global__ __launch_bounds__(256) void gemm_kernel(const TA* __restrict__ A,
                                                   const float* __restrict__ W,
                                                   void* __restrict__ outp) {
  __shared__ _Float16 lA[128 * 40];  // row pitch 40 halves: conflict-free b128
  __shared__ _Float16 lB[128 * 40];

  const int m0 = blockIdx.y * 128;
  const int n0 = blockIdx.x * 128;
  const int t = threadIdx.x;
  const int wave = t >> 5, lane = t & 31;
  const int wm = wave >> 2, wn = wave & 3;      // 2(M) x 4(N) wave grid
  const int lrow = t >> 1, lk = (t & 1) << 4;   // staging: 2 threads/row

  v8f acc[4][2];
#pragma unroll
  for (int i = 0; i < 4; ++i)
#pragma unroll
    for (int j = 0; j < 2; ++j)
#pragma unroll
      for (int e = 0; e < 8; ++e) acc[i][j][e] = 0.0f;

  for (int k0 = 0; k0 < D_MODEL; k0 += 32) {
    if (k0 + 32 < D_MODEL) {  // global_prefetch_b8 for next K-step
      __builtin_prefetch(A + (size_t)(m0 + lrow) * D_MODEL + k0 + 32 + lk, 0, 3);
      __builtin_prefetch(W + (size_t)(n0 + lrow) * D_MODEL + k0 + 32 + lk, 0, 3);
    }
    __syncthreads();
    if constexpr (sizeof(TA) == 2) {  // f16 input: async byte-copy to LDS
      copy32_g2l((const _Float16*)(A + (size_t)(m0 + lrow) * D_MODEL + k0 + lk),
                 lA + lrow * 40 + lk);
    } else {
      stage16(A + (size_t)(m0 + lrow) * D_MODEL + k0 + lk, lA + lrow * 40 + lk);
    }
    stage16(W + (size_t)(n0 + lrow) * D_MODEL + k0 + lk, lB + lrow * 40 + lk);
#if HAS_ASYNC_LDS
    if constexpr (sizeof(TA) == 2) async_wait0();
#endif
    __syncthreads();

    v16h bf[2];
#pragma unroll
    for (int bn = 0; bn < 2; ++bn)
      bf[bn] = frag_b16(lB + (wn * 32 + bn * 16) * 40, 40);
#pragma unroll
    for (int am = 0; am < 4; ++am) {
      v16h af = frag_a16(lA + (wm * 64 + am * 16) * 40, 40);
#pragma unroll
      for (int bn = 0; bn < 2; ++bn)
        acc[am][bn] = wmma_f16(af, bf[bn], acc[am][bn]);
    }
  }

  // Epilogue. C-layout: element r of lane l is (m = r + 8*(l>=16), n = l&15).
  const int half = lane >> 4, nl = lane & 15;
#pragma unroll
  for (int am = 0; am < 4; ++am) {
#pragma unroll
    for (int bn = 0; bn < 2; ++bn) {
      const int gn = n0 + wn * 32 + bn * 16 + nl;
#pragma unroll
      for (int r = 0; r < 8; ++r) {
        const int gm = m0 + wm * 64 + am * 16 + r + half * 8;
        const float v = acc[am][bn][r];
        if (MODE == 2) {
          ((float*)outp)[(size_t)gm * D_MODEL + gn] = v;
        } else {
          const int b = gm >> 11, s = gm & (SEQ - 1);
          const int h = gn >> 6,  d = gn & (DK - 1);
          if (MODE == 0) {
            // RoPE: pair partner (d^1) lives in lane^1 (n = lane&15).
            const float part = __shfl_xor(v, 1, 32);
            const int i2 = d >> 1;
            // inv_freq = 10000^(-2*i/64) = exp(-i * ln(10000)/32)
            const float ang = (float)s * __expf(-0.28782313662425572f * (float)i2);
            const float cs = __cosf(ang), sn = __sinf(ang);
            const float ov = (d & 1) ? (part * sn + v * cs)   // ro = te*sin + to*cos
                                     : (v * cs - part * sn);  // re = te*cos - to*sin
            ((_Float16*)outp)[(((size_t)b * NHEAD + h) * SEQ + s) * DK + d] =
                (_Float16)ov;
          } else {  // MODE 1: V stored transposed [b,h,d,s]
            ((_Float16*)outp)[(((size_t)b * NHEAD + h) * DK + d) * SEQ + s] =
                (_Float16)v;
          }
        }
      }
    }
  }
}

// ---------------------------------------------------------------------------
// Flash attention: one block per (b*h, 128-query tile); 8 waves, 16 rows each.
// Dynamic LDS: K tile [128][72] + V^T tile [64][136] + 8 per-wave P slices
// [16][136] = 70,656 bytes (4 blocks / 320KB WGP).
// ---------------------------------------------------------------------------
#define LDS_K_OFF 0
#define LDS_V_OFF (128 * 72)                 // 9216 halves
#define LDS_P_OFF (LDS_V_OFF + 64 * 136)     // + 8704 halves
#define LDS_TOTAL (LDS_P_OFF + 8 * 16 * 136) // + 17408 -> 35328 halves

__global__ __launch_bounds__(256) void attn_kernel(const _Float16* __restrict__ Qh,
                                                   const _Float16* __restrict__ Kh,
                                                   const _Float16* __restrict__ Vt,
                                                   _Float16* __restrict__ ctx) {
  extern __shared__ _Float16 smem[];
  _Float16* lK = smem + LDS_K_OFF;   // [key][d],  pitch 72
  _Float16* lV = smem + LDS_V_OFF;   // [d][key],  pitch 136
  _Float16* lP = smem + LDS_P_OFF;   // per wave [row][key], pitch 136

  const int bh = blockIdx.y;
  const int q0 = blockIdx.x * 128;
  const _Float16* Qb = Qh + (size_t)bh * SEQ * DK;
  const _Float16* Kb = Kh + (size_t)bh * SEQ * DK;
  const _Float16* Vb = Vt + (size_t)bh * DK * SEQ;   // [d][s]

  const int t = threadIdx.x, wave = t >> 5, lane = t & 31;
  const int half = lane >> 4, nl = lane & 15;

  // Q fragments for this wave's 16 rows, pre-scaled by 1/sqrt(64) = 0.125.
  v16h qf[2];
#pragma unroll
  for (int c = 0; c < 2; ++c) {
    const _Float16* p =
        Qb + (size_t)(q0 + wave * 16 + nl) * DK + c * 32 + (half << 3);
    v8h lo = *(const v8h*)p;
    v8h hi = *(const v8h*)(p + 16);
#pragma unroll
    for (int i = 0; i < 8; ++i) {
      qf[c][i]     = lo[i] * (_Float16)0.125f;
      qf[c][i + 8] = hi[i] * (_Float16)0.125f;
    }
  }

  v8f o[4];
  float mrow[8], lrow[8];
#pragma unroll
  for (int dt = 0; dt < 4; ++dt)
#pragma unroll
    for (int e = 0; e < 8; ++e) o[dt][e] = 0.0f;
#pragma unroll
  for (int r = 0; r < 8; ++r) { mrow[r] = -1e30f; lrow[r] = 0.0f; }

  const int kkey = t >> 1, kdh = (t & 1) << 5;   // K-tile staging map
  const int vd = t >> 2,  vkc = (t & 3) << 5;    // V-tile staging map

  for (int k0 = 0; k0 <= q0; k0 += 128) {
    __syncthreads();
    // Stage K tile (128 keys x 64 d) and V^T tile (64 d x 128 keys): async.
    copy64_g2l(Kb + (size_t)(k0 + kkey) * DK + kdh, lK + kkey * 72 + kdh);
    copy64_g2l(Vb + (size_t)vd * SEQ + k0 + vkc,   lV + vd * 136 + vkc);
    if (k0 + 128 <= q0) {
      __builtin_prefetch(Kb + (size_t)(k0 + 128 + kkey) * DK + kdh, 0, 3);
      __builtin_prefetch(Vb + (size_t)vd * SEQ + k0 + 128 + vkc, 0, 3);
    }
#if HAS_ASYNC_LDS
    async_wait0();
#endif
    __syncthreads();

    // S = (Q/8) * K^T : 8 key-tiles x 2 d-chunks of WMMA
    v8f s[8];
#pragma unroll
    for (int nt = 0; nt < 8; ++nt) {
#pragma unroll
      for (int e = 0; e < 8; ++e) s[nt][e] = 0.0f;
#pragma unroll
      for (int c = 0; c < 2; ++c)
        s[nt] = wmma_f16(qf[c], frag_b16(lK + (nt * 16) * 72 + c * 32, 72),
                         s[nt]);
    }

    if (k0 >= q0) {  // diagonal block: causal mask
#pragma unroll
      for (int nt = 0; nt < 8; ++nt)
#pragma unroll
        for (int r = 0; r < 8; ++r) {
          const int kj = k0 + nt * 16 + nl;
          const int qi = q0 + wave * 16 + r + half * 8;
          if (kj > qi) s[nt][r] = -1e30f;
        }
    }

    // Online softmax (row stats replicated across the 16 lanes of a half).
#pragma unroll
    for (int r = 0; r < 8; ++r) {
      float rm = s[0][r];
#pragma unroll
      for (int nt = 1; nt < 8; ++nt) rm = fmaxf(rm, s[nt][r]);
#pragma unroll
      for (int off = 1; off < 16; off <<= 1)
        rm = fmaxf(rm, __shfl_xor(rm, off, 32));
      const float mnew = fmaxf(mrow[r], rm);
      const float f = __expf(mrow[r] - mnew);
      mrow[r] = mnew;
      lrow[r] *= f;
#pragma unroll
      for (int dt = 0; dt < 4; ++dt) o[dt][r] *= f;
      float rs = 0.0f;
#pragma unroll
      for (int nt = 0; nt < 8; ++nt) {
        const float p = __expf(s[nt][r] - mnew);
        s[nt][r] = p;
        rs += p;
      }
#pragma unroll
      for (int off = 1; off < 16; off <<= 1) rs += __shfl_xor(rs, off, 32);
      lrow[r] += rs;
    }

    // Relayout P: C-layout -> A-fragments via this wave's private LDS slice.
    _Float16* Pw = lP + wave * (16 * 136);
#pragma unroll
    for (int nt = 0; nt < 8; ++nt)
#pragma unroll
      for (int r = 0; r < 8; ++r)
        Pw[(r + half * 8) * 136 + nt * 16 + nl] = (_Float16)s[nt][r];
    asm volatile("s_wait_dscnt 0" ::: "memory");  // LDS stores visible to reads

    // O += P * V  (both operands from LDS)
#pragma unroll
    for (int kc = 0; kc < 4; ++kc) {
      v16h pf = frag_a16(Pw + kc * 32, 136);
#pragma unroll
      for (int dt = 0; dt < 4; ++dt)
        o[dt] = wmma_f16(pf, frag_b16(lV + (dt * 16) * 136 + kc * 32, 136),
                         o[dt]);
    }
  }

  // Normalize and write ctx in [b, s, h*64+d] layout for the final GEMM.
  const int b = bh >> 4, h = bh & 15;
#pragma unroll
  for (int dt = 0; dt < 4; ++dt)
#pragma unroll
    for (int r = 0; r < 8; ++r) {
      const int qi = q0 + wave * 16 + r + half * 8;
      const int d = dt * 16 + nl;
      const float val = o[dt][r] / lrow[r];
      ctx[((size_t)b * SEQ + qi) * D_MODEL + h * DK + d] = (_Float16)val;
    }
}

// ---------------------------------------------------------------------------
extern "C" void kernel_launch(void* const* d_in, const int* in_sizes, int n_in,
                              void* d_out, int out_size, void* d_ws,
                              size_t ws_size, hipStream_t stream) {
  (void)in_sizes; (void)n_in; (void)out_size; (void)ws_size;
  const float* x  = (const float*)d_in[0];
  const float* Wq = (const float*)d_in[1];
  const float* Wk = (const float*)d_in[2];
  const float* Wv = (const float*)d_in[3];
  const float* Wo = (const float*)d_in[4];

  const size_t HE = (size_t)BATCH * NHEAD * SEQ * DK;  // 8,388,608 halves
  _Float16* qh = (_Float16*)d_ws;   // [b,h,s,d]   16 MB
  _Float16* kh = qh + HE;           // [b,h,s,d]   16 MB
  _Float16* vt = kh + HE;           // [b,h,d,s]   16 MB
  _Float16* cx = vt + HE;           // [b,s,1024]  16 MB

  dim3 gridP(D_MODEL / 128, MTOT / 128);  // (8, 64)
  gemm_kernel<0, float><<<gridP, 256, 0, stream>>>(x, Wq, qh);
  gemm_kernel<0, float><<<gridP, 256, 0, stream>>>(x, Wk, kh);
  gemm_kernel<1, float><<<gridP, 256, 0, stream>>>(x, Wv, vt);
  attn_kernel<<<dim3(SEQ / 128, BATCH * NHEAD), 256,
                LDS_TOTAL * sizeof(_Float16), stream>>>(qh, kh, vt, cx);
  gemm_kernel<2, _Float16><<<gridP, 256, 0, stream>>>(cx, Wo, d_out);
}
